// FactormerLayer_90958817394740
// MI455X (gfx1250) — matrix operand order repaired and checked by reading, not compiled, not run on hardware
//
#include <hip/hip_runtime.h>

#define N_NODES 50000
#define E_EDGES 800000
#define DD 64
#define HH 4
#define EFD 96
#define EDIM 32
#define BHID 48
#define FFD 256
#define EUIN 160
#define EUHID 96

typedef __attribute__((ext_vector_type(16))) _Float16 v16h;
typedef __attribute__((ext_vector_type(4)))  _Float16 v4h;
typedef __attribute__((ext_vector_type(8)))  float    v8f;

#define WMMA_F16(a,b,c) __builtin_amdgcn_wmma_f32_16x16x32_f16(false,(a),false,(b),(short)0,(c),false,false)

// ---------------- swizzled fragment layouts (wave32, contiguous k; not ISA table order) ----------------
// A element (m, k): per-lane 16 halves; k%32 = g*16 + h
// swizzled A storage: [kt][lane=g*16+m][h] contiguous -> one v16h load per frag
__device__ __forceinline__ int a_swz_off(int m, int k) {
  int kt = k >> 5, kr = k & 31;
  int g = (kr >> 4) & 1;
  int h = kr & 15;
  return ((kt << 5) + (g << 4) + m) * 16 + h;
}
__device__ __forceinline__ v16h a_frag_swz(const v16h* a, int lane, int kt) {
  return a[kt * 32 + lane];
}
// B element (k, c): lane = g*16 + n, half h; k%32 = g*16 + h
// swizzled B storage: [kt][nt][lane][h]
__device__ __forceinline__ v16h b_frag_swz(const v16h* w, int lane, int kt, int nt, int nnt) {
  return w[(kt * nnt + nt) * 32 + lane];
}
__device__ __forceinline__ void fill_b_swz(v16h* dst, const float* w, int K, int NCOL,
                                           int tid, int nthr) {
  _Float16* d = (_Float16*)dst;
  int nnt = NCOL >> 4;
  for (int i = tid; i < K * NCOL; i += nthr) {
    int k = i / NCOL, c = i % NCOL;
    int kt = k >> 5, g = (k >> 4) & 1, h = k & 15;
    int nt = c >> 4, n = c & 15;
    d[(((kt * nnt + nt) << 5) + (g << 4) + n) * 16 + h] = (_Float16)w[i];
  }
}

__device__ __forceinline__ void atomicMaxFloat(float* addr, float val) {
  if (val >= 0.0f) atomicMax((int*)addr, __float_as_int(val));
  else             atomicMin((unsigned int*)addr, __float_as_uint(val));
}

// layernorm a row of 64 and write f16 into swizzled A tile
__device__ __forceinline__ void ln_row64_swz(const float* src, _Float16* dst, int m,
                                             const float* gg, const float* bb) {
  float mean = 0.f;
  for (int c = 0; c < 64; ++c) mean += src[c];
  mean *= (1.f / 64.f);
  float var = 0.f;
  for (int c = 0; c < 64; ++c) { float d = src[c] - mean; var += d * d; }
  var *= (1.f / 64.f);
  float rinv = rsqrtf(var + 1e-5f);
  for (int c = 0; c < 64; c += 4) {
    v4h p;
#pragma unroll
    for (int j = 0; j < 4; ++j) p[j] = (_Float16)((src[c + j] - mean) * rinv * gg[c + j] + bb[c + j]);
    *(v4h*)(dst + a_swz_off(m, c)) = p;
  }
}

// ================= init =================
__global__ void k_init(float* seg_max, float* seg_sum, float* agg) {
  int idx = blockIdx.x * 256 + threadIdx.x;
  if (idx < N_NODES * DD) agg[idx] = 0.f;
  if (idx < N_NODES * HH) { seg_max[idx] = -__builtin_inff(); seg_sum[idx] = 0.f; }
}

// ================= node QKV (5 waves, 160 thr) =================
__global__ void k_node_qkv(const float* __restrict__ source, const float* __restrict__ target,
                           const float* __restrict__ qw, const float* __restrict__ knw,
                           const float* __restrict__ vnw,
                           const float* __restrict__ nq_g, const float* __restrict__ nq_b,
                           const float* __restrict__ nkv_g, const float* __restrict__ nkv_b,
                           float* __restrict__ Q, float* __restrict__ Kn, float* __restrict__ Vn) {
  __shared__ v16h wQ[256], wK[256], wV[256];   // 2kt x 4nt x 32
  __shared__ float4 stg[5][256];               // 16x64 f32
  __shared__ v16h  a16[5][64];                 // 16x64 f16 swizzled
  int tid = threadIdx.x, lane = tid & 31, wid = tid >> 5;
  fill_b_swz(wQ, qw, 64, 64, tid, 160);
  fill_b_swz(wK, knw, 64, 64, tid, 160);
  fill_b_swz(wV, vnw, 64, 64, tid, 160);
  __syncthreads();
  int r0 = (blockIdx.x * 5 + wid) * 16;
  int n = lane & 15, g = lane >> 4;
  const float4* tg4 = (const float4*)target;
  const float4* sr4 = (const float4*)source;

  for (int i = 0; i < 8; ++i) { int idx = i * 32 + lane; stg[wid][idx] = tg4[r0 * 16 + idx]; }
  __syncthreads();
  if (lane < 16) ln_row64_swz((const float*)stg[wid] + lane * 64, (_Float16*)a16[wid], lane, nq_g, nq_b);
  __syncthreads();
#pragma unroll
  for (int ct = 0; ct < 4; ++ct) {
    v8f c = {};
#pragma unroll
    for (int kt = 0; kt < 2; ++kt)
      c = WMMA_F16(a_frag_swz(a16[wid], lane, kt), b_frag_swz(wQ, lane, kt, ct, 4), c);
#pragma unroll
    for (int vv = 0; vv < 8; ++vv) Q[(r0 + vv + 8 * g) * 64 + ct * 16 + n] = c[vv];
  }
  __syncthreads();
  for (int i = 0; i < 8; ++i) { int idx = i * 32 + lane; stg[wid][idx] = sr4[r0 * 16 + idx]; }
  __syncthreads();
  if (lane < 16) ln_row64_swz((const float*)stg[wid] + lane * 64, (_Float16*)a16[wid], lane, nkv_g, nkv_b);
  __syncthreads();
#pragma unroll
  for (int ct = 0; ct < 4; ++ct) {
    v8f c = {};
#pragma unroll
    for (int kt = 0; kt < 2; ++kt)
      c = WMMA_F16(a_frag_swz(a16[wid], lane, kt), b_frag_swz(wK, lane, kt, ct, 4), c);
#pragma unroll
    for (int vv = 0; vv < 8; ++vv) Kn[(r0 + vv + 8 * g) * 64 + ct * 16 + n] = c[vv];
  }
#pragma unroll
  for (int ct = 0; ct < 4; ++ct) {
    v8f c = {};
#pragma unroll
    for (int kt = 0; kt < 2; ++kt)
      c = WMMA_F16(a_frag_swz(a16[wid], lane, kt), b_frag_swz(wV, lane, kt, ct, 4), c);
#pragma unroll
    for (int vv = 0; vv < 8; ++vv) Vn[(r0 + vv + 8 * g) * 64 + ct * 16 + n] = c[vv];
  }
}

// ================= edge pass 1 (2 waves, 64 thr) =================
__global__ void k_edge1(const float* __restrict__ src_nodes, const float* __restrict__ tgt_nodes,
                        const int* __restrict__ ei, const float* __restrict__ edge_attr,
                        const float* __restrict__ enorm_g, const float* __restrict__ enorm_b,
                        const float* __restrict__ kEw, const float* __restrict__ kEb,
                        const float* __restrict__ vEw, const float* __restrict__ vEb,
                        const float* __restrict__ eb1w, const float* __restrict__ eb1b,
                        const float* __restrict__ eb2w, const float* __restrict__ eb2b,
                        const float* __restrict__ inv_temp,
                        const float* __restrict__ Q, const float* __restrict__ Kn,
                        const float* __restrict__ Vn,
                        float* __restrict__ ws_ef, float* __restrict__ ws_sv,
                        float* __restrict__ ws_logits, float* __restrict__ seg_max) {
  __shared__ v16h wK[384], wV[384];      // 3kt x 4nt x 32
  __shared__ v16h wB1[288];              // 3kt x 3nt x 32
  __shared__ v16h wB2[64];               // 2kt x 1nt x 32 (padded 64x16)
  __shared__ float4 scr4[2][384];        // 16x96 f32
  __shared__ v16h  f16t[2][96];          // 16x96 f16 swizzled
  __shared__ v16h  hid[2][64];           // 16x64 f16 swizzled
  __shared__ float ltile[2][64];
  __shared__ int   sidx[2][16], tidx[2][16];
  int tid = threadIdx.x, lane = tid & 31, wid = tid >> 5;
  fill_b_swz(wK, kEw, 96, 64, tid, 64);
  fill_b_swz(wV, vEw, 96, 64, tid, 64);
  fill_b_swz(wB1, eb1w, 96, 48, tid, 64);
  { v16h z = {}; wB2[tid] = z; }
  __syncthreads();
  for (int i = tid; i < BHID * HH; i += 64) {
    int k = i >> 2, c = i & 3;
    int kt = k >> 5, g = (k >> 4) & 1, h = k & 15;
    ((_Float16*)wB2)[((kt << 5) + (g << 4) + c) * 16 + h] = (_Float16)eb2w[i];
  }
  __syncthreads();

  int e0 = (blockIdx.x * 2 + wid) * 16;
  if (lane < 16) { sidx[wid][lane] = ei[e0 + lane]; tidx[wid][lane] = ei[E_EDGES + e0 + lane]; }
  __syncthreads();

  float* scr = (float*)scr4[wid];
  const float4* s4 = (const float4*)src_nodes;
  const float4* t4 = (const float4*)tgt_nodes;
  const float4* ea4 = (const float4*)edge_attr;
  // edge features: [pairwise(64) | edge_attr(32)] -> scr (16x96 f32)
  for (int i = 0; i < 12; ++i) {
    int idx = i * 32 + lane; int r = idx / 24, c4 = idx % 24;
    float4 v;
    if (c4 < 16) {
      float4 a = s4[sidx[wid][r] * 16 + c4];
      float4 b = t4[tidx[wid][r] * 16 + c4];
      v.x = a.x * b.x; v.y = a.y * b.y; v.z = a.z * b.z; v.w = a.w * b.w;
    } else {
      v = ea4[(e0 + r) * 8 + (c4 - 16)];
    }
    scr4[wid][idx] = v;
  }
  __syncthreads();
  if (lane < 16) {
    float* row = scr + lane * 96;
    float mean = 0.f;
    for (int c = 0; c < 96; ++c) mean += row[c];
    mean *= (1.f / 96.f);
    float var = 0.f;
    for (int c = 0; c < 96; ++c) { float d = row[c] - mean; var += d * d; }
    var *= (1.f / 96.f);
    float rinv = rsqrtf(var + 1e-5f);
    _Float16* ft = (_Float16*)f16t[wid];
    for (int c = 0; c < 96; c += 4) {
      v4h p;
#pragma unroll
      for (int j = 0; j < 4; ++j) {
        float v = (row[c + j] - mean) * rinv * enorm_g[c + j] + enorm_b[c + j];
        row[c + j] = v; p[j] = (_Float16)v;
      }
      *(v4h*)(ft + a_swz_off(lane, c)) = p;
    }
  }
  __syncthreads();
  {
    float4* ef4 = (float4*)(ws_ef + (size_t)e0 * 96);
    for (int i = 0; i < 12; ++i) { int idx = i * 32 + lane; ef4[idx] = scr4[wid][idx]; }
  }

  int n = lane & 15, g = lane >> 4;
  // K_edge + bias
  v8f cK[4];
#pragma unroll
  for (int ct = 0; ct < 4; ++ct) {
    v8f c = {};
#pragma unroll
    for (int kt = 0; kt < 3; ++kt)
      c = WMMA_F16(a_frag_swz(f16t[wid], lane, kt), b_frag_swz(wK, lane, kt, ct, 4), c);
    float bb = kEb[ct * 16 + n];
#pragma unroll
    for (int vv = 0; vv < 8; ++vv) c[vv] += bb;
    cK[ct] = c;
  }
  __syncthreads();
  // + Kn[src]
  {
    const float4* kn4 = (const float4*)Kn;
    for (int i = 0; i < 8; ++i) { int idx = i * 32 + lane; scr4[wid][idx] = kn4[sidx[wid][idx >> 4] * 16 + (idx & 15)]; }
  }
  __syncthreads();
#pragma unroll
  for (int ct = 0; ct < 4; ++ct)
#pragma unroll
    for (int vv = 0; vv < 8; ++vv) cK[ct][vv] += scr[(vv + 8 * g) * 64 + ct * 16 + n];
  __syncthreads();
  // logits: per-head dot with Q[tgt]
  {
    const float4* q4 = (const float4*)Q;
    for (int i = 0; i < 8; ++i) { int idx = i * 32 + lane; scr4[wid][idx] = q4[tidx[wid][idx >> 4] * 16 + (idx & 15)]; }
  }
  __syncthreads();
#pragma unroll
  for (int ct = 0; ct < 4; ++ct) {
    float it = inv_temp[ct];
#pragma unroll
    for (int vv = 0; vv < 8; ++vv) {
      float p = scr[(vv + 8 * g) * 64 + ct * 16 + n] * cK[ct][vv];
      p += __shfl_xor(p, 1); p += __shfl_xor(p, 2); p += __shfl_xor(p, 4); p += __shfl_xor(p, 8);
      if (n == 0) ltile[wid][(vv + 8 * g) * 4 + ct] = p * it;
    }
  }
  __syncthreads();
  // V_edge + Vn[src] -> sum_vals
  v8f cV[4];
#pragma unroll
  for (int ct = 0; ct < 4; ++ct) {
    v8f c = {};
#pragma unroll
    for (int kt = 0; kt < 3; ++kt)
      c = WMMA_F16(a_frag_swz(f16t[wid], lane, kt), b_frag_swz(wV, lane, kt, ct, 4), c);
    float bb = vEb[ct * 16 + n];
#pragma unroll
    for (int vv = 0; vv < 8; ++vv) c[vv] += bb;
    cV[ct] = c;
  }
  {
    const float4* vn4 = (const float4*)Vn;
    for (int i = 0; i < 8; ++i) { int idx = i * 32 + lane; scr4[wid][idx] = vn4[sidx[wid][idx >> 4] * 16 + (idx & 15)]; }
  }
  __syncthreads();
#pragma unroll
  for (int ct = 0; ct < 4; ++ct)
#pragma unroll
    for (int vv = 0; vv < 8; ++vv) {
      float sv = cV[ct][vv] + scr[(vv + 8 * g) * 64 + ct * 16 + n];
      ws_sv[(size_t)(e0 + vv + 8 * g) * 64 + ct * 16 + n] = sv;
    }
  // attention bias MLP: relu(ef@eb1+b1)@eb2+b2
  { v16h z = {}; hid[wid][lane] = z; hid[wid][lane + 32] = z; }
  __syncthreads();
  _Float16* hp = (_Float16*)hid[wid];
#pragma unroll
  for (int ct = 0; ct < 3; ++ct) {
    v8f c = {};
#pragma unroll
    for (int kt = 0; kt < 3; ++kt)
      c = WMMA_F16(a_frag_swz(f16t[wid], lane, kt), b_frag_swz(wB1, lane, kt, ct, 3), c);
    float bb = eb1b[ct * 16 + n];
#pragma unroll
    for (int vv = 0; vv < 8; ++vv)
      hp[a_swz_off(vv + 8 * g, ct * 16 + n)] = (_Float16)fmaxf(c[vv] + bb, 0.f);
  }
  __syncthreads();
  v8f c2 = {};
#pragma unroll
  for (int kt = 0; kt < 2; ++kt)
    c2 = WMMA_F16(a_frag_swz(hid[wid], lane, kt), b_frag_swz(wB2, lane, kt, 0, 1), c2);
  if (n < 4) {
#pragma unroll
    for (int vv = 0; vv < 8; ++vv) {
      int m = vv + 8 * g;
      float lg = ltile[wid][m * 4 + n] + c2[vv] + eb2b[n];
      ws_logits[(size_t)(e0 + m) * 4 + n] = lg;
      atomicMaxFloat(&seg_max[tidx[wid][m] * 4 + n], lg);
    }
  }
}

// ================= edge pass 2 + edge update (2 waves, 64 thr) =================
__global__ void k_edge2(const int* __restrict__ ei, const float* __restrict__ ws_logits,
                        const float* __restrict__ seg_max,
                        const float* __restrict__ ws_ef, const float* __restrict__ ws_sv,
                        const float* __restrict__ eu1w, const float* __restrict__ eu1b,
                        const float* __restrict__ eu2w, const float* __restrict__ eu2b,
                        float* __restrict__ seg_sum, float* __restrict__ agg,
                        float* __restrict__ edge_out) {
  __shared__ v16h W1[960];               // 5kt x 6nt x 32
  __shared__ v16h W2[192];               // 3kt x 2nt x 32
  __shared__ v16h a16[2][160];           // 16x160 swizzled
  __shared__ v16h hid[2][96];            // 16x96 swizzled
  __shared__ float zt[2][64];
  __shared__ int   tix[2][16];
  int tid = threadIdx.x, lane = tid & 31, wid = tid >> 5;
  fill_b_swz(W1, eu1w, 160, 96, tid, 64);
  fill_b_swz(W2, eu2w, 96, 32, tid, 64);

  int e0 = (blockIdx.x * 2 + wid) * 16;
  if (lane < 16) {
    int t = ei[E_EDGES + e0 + lane]; tix[wid][lane] = t;
    for (int h = 0; h < 4; ++h) {
      float l = ws_logits[(size_t)(e0 + lane) * 4 + h];
      float m = seg_max[t * 4 + h];
      float z = expf(l - m);
      zt[wid][lane * 4 + h] = z;
      atomicAdd(&seg_sum[t * 4 + h], z);
    }
  }
  __syncthreads();
  const float4* sv4 = (const float4*)ws_sv;
  for (int i = 0; i < 8; ++i) {
    int idx = i * 32 + lane; int r = idx >> 4, c4 = idx & 15;
    float z = zt[wid][r * 4 + (c4 >> 2)];
    float4 sv = sv4[(size_t)e0 * 16 + idx];
    float* ap = &agg[tix[wid][r] * 64 + c4 * 4];
    atomicAdd(ap + 0, z * sv.x); atomicAdd(ap + 1, z * sv.y);
    atomicAdd(ap + 2, z * sv.z); atomicAdd(ap + 3, z * sv.w);
  }
  // edge-update MLP input: [edge_feat(96) | sum_vals(64)] -> swizzled A
  {
    const float4* ef4 = (const float4*)ws_ef;
    _Float16* ap = (_Float16*)a16[wid];
    for (int i = 0; i < 20; ++i) {
      int idx = i * 32 + lane; int r = idx / 40, c4 = idx % 40;
      float4 v = (c4 < 24) ? ef4[(size_t)(e0 + r) * 24 + c4]
                           : sv4[(size_t)(e0 + r) * 16 + (c4 - 24)];
      v4h p; p[0] = (_Float16)v.x; p[1] = (_Float16)v.y; p[2] = (_Float16)v.z; p[3] = (_Float16)v.w;
      *(v4h*)(ap + a_swz_off(r, c4 * 4)) = p;
    }
  }
  __syncthreads();
  int n = lane & 15, g = lane >> 4;
  _Float16* hp = (_Float16*)hid[wid];
#pragma unroll
  for (int ct = 0; ct < 6; ++ct) {
    v8f c = {};
#pragma unroll
    for (int kt = 0; kt < 5; ++kt)
      c = WMMA_F16(a_frag_swz(a16[wid], lane, kt), b_frag_swz(W1, lane, kt, ct, 6), c);
    float bb = eu1b[ct * 16 + n];
#pragma unroll
    for (int vv = 0; vv < 8; ++vv)
      hp[a_swz_off(vv + 8 * g, ct * 16 + n)] = (_Float16)fmaxf(c[vv] + bb, 0.f);
  }
  __syncthreads();
#pragma unroll
  for (int ct = 0; ct < 2; ++ct) {
    v8f c = {};
#pragma unroll
    for (int kt = 0; kt < 3; ++kt)
      c = WMMA_F16(a_frag_swz(hid[wid], lane, kt), b_frag_swz(W2, lane, kt, ct, 2), c);
    float bb = eu2b[ct * 16 + n];
#pragma unroll
    for (int vv = 0; vv < 8; ++vv)
      edge_out[(size_t)(e0 + vv + 8 * g) * 32 + ct * 16 + n] = c[vv] + bb;
  }
}

// ================= attention out + residual + LN1 (5 waves) =================
__global__ void k_attnout_ln1(const float* __restrict__ agg, const float* __restrict__ segsum,
                              const float* __restrict__ target,
                              const float* __restrict__ out_w, const float* __restrict__ out_b,
                              const float* __restrict__ ln1_g, const float* __restrict__ ln1_b,
                              const float* __restrict__ res_scale,
                              float* __restrict__ y_ws, float* __restrict__ y1_ws) {
  __shared__ v16h wO[256];
  __shared__ v16h a16[5][64];
  __shared__ float ot[5][1024];
  int tid = threadIdx.x, lane = tid & 31, wid = tid >> 5;
  fill_b_swz(wO, out_w, 64, 64, tid, 160);
  __syncthreads();
  int r0 = (blockIdx.x * 5 + wid) * 16;
  {
    const float4* ag4 = (const float4*)agg;
    _Float16* ap = (_Float16*)a16[wid];
    for (int i = 0; i < 8; ++i) {
      int idx = i * 32 + lane; int r = idx >> 4, c4 = idx & 15;
      float den = segsum[(r0 + r) * 4 + (c4 >> 2)];
      float4 a = ag4[(size_t)(r0 + r) * 16 + c4];
      float rd = (den > 0.f) ? 1.f / den : 0.f;
      v4h p; p[0] = (_Float16)(a.x * rd); p[1] = (_Float16)(a.y * rd);
      p[2] = (_Float16)(a.z * rd); p[3] = (_Float16)(a.w * rd);
      *(v4h*)(ap + a_swz_off(r, c4 * 4)) = p;
    }
  }
  __syncthreads();
  int n = lane & 15, g = lane >> 4;
#pragma unroll
  for (int ct = 0; ct < 4; ++ct) {
    v8f c = {};
#pragma unroll
    for (int kt = 0; kt < 2; ++kt)
      c = WMMA_F16(a_frag_swz(a16[wid], lane, kt), b_frag_swz(wO, lane, kt, ct, 4), c);
    float bb = out_b[ct * 16 + n];
#pragma unroll
    for (int vv = 0; vv < 8; ++vv) ot[wid][(vv + 8 * g) * 64 + ct * 16 + n] = c[vv] + bb;
  }
  __syncthreads();
  float rs = res_scale[0];
  if (lane < 16) {
    int row = r0 + lane;
    float* rt = ot[wid] + lane * 64;
    float mean = 0.f;
    for (int c = 0; c < 64; ++c) { float yv = target[row * 64 + c] + rs * rt[c]; rt[c] = yv; mean += yv; }
    mean *= (1.f / 64.f);
    float var = 0.f;
    for (int c = 0; c < 64; ++c) { float d = rt[c] - mean; var += d * d; }
    var *= (1.f / 64.f);
    float rinv = rsqrtf(var + 1e-5f);
    for (int c = 0; c < 64; ++c) {
      y_ws[row * 64 + c] = rt[c];
      y1_ws[row * 64 + c] = (rt[c] - mean) * rinv * ln1_g[c] + ln1_b[c];
    }
  }
}

// ================= FFN up: gelu(y1@Wu+b) -> h_ws =================
__global__ void k_ffn_up(const float* __restrict__ y1_ws, const float* __restrict__ w,
                         const float* __restrict__ bias, float* __restrict__ h_ws) {
  __shared__ v16h W[1024];   // 2kt x 16nt x 32
  __shared__ v16h a16[5][64];
  int tid = threadIdx.x, lane = tid & 31, wid = tid >> 5;
  fill_b_swz(W, w, 64, 256, tid, 160);
  __syncthreads();
  int r0 = (blockIdx.x * 5 + wid) * 16;
  {
    const float4* y4 = (const float4*)y1_ws;
    _Float16* ap = (_Float16*)a16[wid];
    for (int i = 0; i < 8; ++i) {
      int idx = i * 32 + lane; int r = idx >> 4, c4 = idx & 15;
      float4 v = y4[(size_t)(r0 + r) * 16 + c4];
      v4h p; p[0] = (_Float16)v.x; p[1] = (_Float16)v.y; p[2] = (_Float16)v.z; p[3] = (_Float16)v.w;
      *(v4h*)(ap + a_swz_off(r, c4 * 4)) = p;
    }
  }
  __syncthreads();
  int n = lane & 15, g = lane >> 4;
#pragma unroll
  for (int ct = 0; ct < 16; ++ct) {
    v8f c = {};
#pragma unroll
    for (int kt = 0; kt < 2; ++kt)
      c = WMMA_F16(a_frag_swz(a16[wid], lane, kt), b_frag_swz(W, lane, kt, ct, 16), c);
    float bb = bias[ct * 16 + n];
#pragma unroll
    for (int vv = 0; vv < 8; ++vv) {
      float x = c[vv] + bb;
      h_ws[(size_t)(r0 + vv + 8 * g) * 256 + ct * 16 + n] = 0.5f * x * (1.f + erff(x * 0.70710678118654752f));
    }
  }
}

// ================= FFN gate: h_ws *= (y1@Wg+b) =================
__global__ void k_ffn_gate(const float* __restrict__ y1_ws, const float* __restrict__ w,
                           const float* __restrict__ bias, float* __restrict__ h_ws) {
  __shared__ v16h W[1024];
  __shared__ v16h a16[5][64];
  int tid = threadIdx.x, lane = tid & 31, wid = tid >> 5;
  fill_b_swz(W, w, 64, 256, tid, 160);
  __syncthreads();
  int r0 = (blockIdx.x * 5 + wid) * 16;
  {
    const float4* y4 = (const float4*)y1_ws;
    _Float16* ap = (_Float16*)a16[wid];
    for (int i = 0; i < 8; ++i) {
      int idx = i * 32 + lane; int r = idx >> 4, c4 = idx & 15;
      float4 v = y4[(size_t)(r0 + r) * 16 + c4];
      v4h p; p[0] = (_Float16)v.x; p[1] = (_Float16)v.y; p[2] = (_Float16)v.z; p[3] = (_Float16)v.w;
      *(v4h*)(ap + a_swz_off(r, c4 * 4)) = p;
    }
  }
  __syncthreads();
  int n = lane & 15, g = lane >> 4;
#pragma unroll
  for (int ct = 0; ct < 16; ++ct) {
    v8f c = {};
#pragma unroll
    for (int kt = 0; kt < 2; ++kt)
      c = WMMA_F16(a_frag_swz(a16[wid], lane, kt), b_frag_swz(W, lane, kt, ct, 16), c);
    float bb = bias[ct * 16 + n];
#pragma unroll
    for (int vv = 0; vv < 8; ++vv) {
      size_t o = (size_t)(r0 + vv + 8 * g) * 256 + ct * 16 + n;
      h_ws[o] = (c[vv] + bb) * h_ws[o];
    }
  }
}

// ================= FFN down + residual + LN2 -> d_out =================
__global__ void k_ffn_down_ln2(const float* __restrict__ h_ws, const float* __restrict__ w,
                               const float* __restrict__ bias, const float* __restrict__ y_ws,
                               const float* __restrict__ ffn_scale,
                               const float* __restrict__ ln2_g, const float* __restrict__ ln2_b,
                               float* __restrict__ out) {
  __shared__ v16h W[1024];     // 8kt x 4nt x 32
  __shared__ v16h a16c[5][32]; // 16x32 swizzled chunk
  __shared__ float ot[5][1024];
  int tid = threadIdx.x, lane = tid & 31, wid = tid >> 5;
  fill_b_swz(W, w, 256, 64, tid, 160);
  __syncthreads();
  int r0 = (blockIdx.x * 5 + wid) * 16;
  int n = lane & 15, g = lane >> 4;
  const float4* h4 = (const float4*)h_ws;
  v8f acc[4] = {};
  for (int kc = 0; kc < 8; ++kc) {
    _Float16* ap = (_Float16*)a16c[wid];
    for (int i = 0; i < 4; ++i) {
      int idx = i * 32 + lane; int r = idx >> 3, c4 = idx & 7;
      float4 v = h4[(size_t)(r0 + r) * 64 + kc * 8 + c4];
      v4h p; p[0] = (_Float16)v.x; p[1] = (_Float16)v.y; p[2] = (_Float16)v.z; p[3] = (_Float16)v.w;
      *(v4h*)(ap + a_swz_off(r, c4 * 4)) = p;
    }
    __syncthreads();
#pragma unroll
    for (int ct = 0; ct < 4; ++ct)
      acc[ct] = WMMA_F16(a_frag_swz(a16c[wid], lane, 0), b_frag_swz(W, lane, kc, ct, 4), acc[ct]);
    __syncthreads();
  }
#pragma unroll
  for (int ct = 0; ct < 4; ++ct) {
    float bb = bias[ct * 16 + n];
#pragma unroll
    for (int vv = 0; vv < 8; ++vv) ot[wid][(vv + 8 * g) * 64 + ct * 16 + n] = acc[ct][vv] + bb;
  }
  __syncthreads();
  float fs = ffn_scale[0];
  if (lane < 16) {
    int row = r0 + lane;
    float* rt = ot[wid] + lane * 64;
    float mean = 0.f;
    for (int c = 0; c < 64; ++c) { float yv = y_ws[row * 64 + c] + fs * rt[c]; rt[c] = yv; mean += yv; }
    mean *= (1.f / 64.f);
    float var = 0.f;
    for (int c = 0; c < 64; ++c) { float d = rt[c] - mean; var += d * d; }
    var *= (1.f / 64.f);
    float rinv = rsqrtf(var + 1e-5f);
    for (int c = 0; c < 64; ++c)
      out[row * 64 + c] = (rt[c] - mean) * rinv * ln2_g[c] + ln2_b[c];
  }
}

extern "C" void kernel_launch(void* const* d_in, const int* in_sizes, int n_in,
                              void* d_out, int out_size, void* d_ws, size_t ws_size,
                              hipStream_t stream) {
  const float* source    = (const float*)d_in[0];
  const float* target    = (const float*)d_in[1];
  const int*   edge_index= (const int*)  d_in[2];
  const float* edge_attr = (const float*)d_in[3];
  const float* q_w   = (const float*)d_in[4];
  const float* kN_w  = (const float*)d_in[5];
  const float* vN_w  = (const float*)d_in[6];
  const float* kE_w  = (const float*)d_in[7];
  const float* kE_b  = (const float*)d_in[8];
  const float* vE_w  = (const float*)d_in[9];
  const float* vE_b  = (const float*)d_in[10];
  const float* eb1_w = (const float*)d_in[11];
  const float* eb1_b = (const float*)d_in[12];
  const float* eb2_w = (const float*)d_in[13];
  const float* eb2_b = (const float*)d_in[14];
  const float* en_g  = (const float*)d_in[15];
  const float* en_b  = (const float*)d_in[16];
  const float* out_w = (const float*)d_in[17];
  const float* out_b = (const float*)d_in[18];
  const float* fg_w  = (const float*)d_in[19];
  const float* fg_b  = (const float*)d_in[20];
  const float* fu_w  = (const float*)d_in[21];
  const float* fu_b  = (const float*)d_in[22];
  const float* fd_w  = (const float*)d_in[23];
  const float* fd_b  = (const float*)d_in[24];
  const float* nq_g  = (const float*)d_in[25];
  const float* nq_b  = (const float*)d_in[26];
  const float* nkv_g = (const float*)d_in[27];
  const float* nkv_b = (const float*)d_in[28];
  const float* ln1_g = (const float*)d_in[29];
  const float* ln1_b = (const float*)d_in[30];
  const float* ln2_g = (const float*)d_in[31];
  const float* ln2_b = (const float*)d_in[32];
  const float* res_s = (const float*)d_in[33];
  const float* ffn_s = (const float*)d_in[34];
  const float* invt  = (const float*)d_in[35];
  const float* eu1_w = (const float*)d_in[36];
  const float* eu1_b = (const float*)d_in[37];
  const float* eu2_w = (const float*)d_in[38];
  const float* eu2_b = (const float*)d_in[39];

  float* ws = (float*)d_ws;
  size_t off = 0;
  float* Q      = ws + off; off += (size_t)N_NODES * DD;
  float* Kn     = ws + off; off += (size_t)N_NODES * DD;
  float* Vn     = ws + off; off += (size_t)N_NODES * DD;
  float* ws_ef  = ws + off; off += (size_t)E_EDGES * EFD;
  float* ws_sv  = ws + off; off += (size_t)E_EDGES * DD;
  float* ws_lg  = ws + off; off += (size_t)E_EDGES * HH;
  float* segmax = ws + off; off += (size_t)N_NODES * HH;
  float* segsum = ws + off; off += (size_t)N_NODES * HH;
  float* agg    = ws + off; off += (size_t)N_NODES * DD;
  float* y_ws   = ws + off; off += (size_t)N_NODES * DD;
  float* y1_ws  = ws + off; off += (size_t)N_NODES * DD;
  float* h_ws   = ws + off; off += (size_t)N_NODES * FFD;

  float* y_out    = (float*)d_out;
  float* edge_out = (float*)d_out + (size_t)N_NODES * DD;

  k_init<<<(N_NODES * DD + 255) / 256, 256, 0, stream>>>(segmax, segsum, agg);
  k_node_qkv<<<625, 160, 0, stream>>>(source, target, q_w, kN_w, vN_w,
                                      nq_g, nq_b, nkv_g, nkv_b, Q, Kn, Vn);
  k_edge1<<<25000, 64, 0, stream>>>(source, target, edge_index, edge_attr, en_g, en_b,
                                    kE_w, kE_b, vE_w, vE_b, eb1_w, eb1_b, eb2_w, eb2_b,
                                    invt, Q, Kn, Vn, ws_ef, ws_sv, ws_lg, segmax);
  k_edge2<<<25000, 64, 0, stream>>>(edge_index, ws_lg, segmax, ws_ef, ws_sv,
                                    eu1_w, eu1_b, eu2_w, eu2_b, segsum, agg, edge_out);
  k_attnout_ln1<<<625, 160, 0, stream>>>(agg, segsum, target, out_w, out_b,
                                         ln1_g, ln1_b, res_s, y_ws, y1_ws);
  k_ffn_up<<<625, 160, 0, stream>>>(y1_ws, fu_w, fu_b, h_ws);
  k_ffn_gate<<<625, 160, 0, stream>>>(y1_ws, fg_w, fg_b, h_ws);
  k_ffn_down_ln2<<<625, 160, 0, stream>>>(h_ws, fd_w, fd_b, y_ws, ffn_s, ln2_g, ln2_b, y_out);
}
